// HopfieldMemoryLayer_20744692039862
// MI455X (gfx1250) — compile-verified, matrix operand order checked by reading, not run on hardware
//
#include <hip/hip_runtime.h>
#include <hip/hip_bf16.h>

// ---------------------------------------------------------------------------
// Hopfield memory layer on MI455X (gfx1250), wave32 + v_wmma_f32_16x16x32_bf16
// ---------------------------------------------------------------------------

typedef __attribute__((ext_vector_type(16))) __bf16 v16bf;
typedef __attribute__((ext_vector_type(8)))  float  v8f;

struct Frag32B { uint4 lo, hi; };

// Load one 16-lane-striped WMMA operand fragment from a UNIFORM base pointer
// plus a 32-bit per-lane element offset (selects GVS: saddr + 32-bit vaddr).
// Each lane supplies two contiguous 16-byte chunks (K=off..off+7, 16+off..).
__device__ __forceinline__ v16bf load_frag_u(const __hip_bfloat16* base,
                                             unsigned off) {
  Frag32B f;
  f.lo = *reinterpret_cast<const uint4*>(base + off);
  f.hi = *reinterpret_cast<const uint4*>(base + off + 16u);
  return __builtin_bit_cast(v16bf, f);
}

// Generic-pointer variant (used for the LDS P scratch).
__device__ __forceinline__ v16bf load_frag(const __hip_bfloat16* p0,
                                           const __hip_bfloat16* p1) {
  Frag32B f;
  f.lo = *reinterpret_cast<const uint4*>(p0);
  f.hi = *reinterpret_cast<const uint4*>(p1);
  return __builtin_bit_cast(v16bf, f);
}

__device__ __forceinline__ v8f zero8() {
  v8f z;
#pragma unroll
  for (int i = 0; i < 8; ++i) z[i] = 0.0f;
  return z;
}

__device__ __forceinline__ v8f wmma_bf16(v16bf a, v16bf b, v8f c) {
  // D = A(16x32 bf16) * B(32x16 bf16) + C(16x16 f32)
  return __builtin_amdgcn_wmma_f32_16x16x32_bf16(
      false, a, false, b, (short)0, c, false, false);
}

// ---------------------------------------------------------------------------
// fp32 -> bf16 cast
// ---------------------------------------------------------------------------
__global__ void cast_bf16_kernel(const float* __restrict__ x,
                                 __hip_bfloat16* __restrict__ y, long n) {
  long i = (long)blockIdx.x * blockDim.x + threadIdx.x;
  if (i < n) y[i] = __float2bfloat16(x[i]);
}

// ---------------------------------------------------------------------------
// RMSNorm over E=2048 + weight + cast to bf16 (one block per row)
// ---------------------------------------------------------------------------
__global__ void rmsnorm_cast_kernel(const float* __restrict__ x,
                                    const float* __restrict__ w,
                                    __hip_bfloat16* __restrict__ y) {
  __shared__ float red[256];
  long row = blockIdx.x;
  const float* xr = x + row * 2048;
  float ss = 0.0f;
  for (int c = threadIdx.x; c < 2048; c += 256) {
    float v = xr[c];
    ss += v * v;
  }
  red[threadIdx.x] = ss;
  __syncthreads();
  for (int s = 128; s > 0; s >>= 1) {
    if ((int)threadIdx.x < s) red[threadIdx.x] += red[threadIdx.x + s];
    __syncthreads();
  }
  float rn = rsqrtf(red[0] * (1.0f / 2048.0f) + 1e-6f);
  __hip_bfloat16* yr = y + row * 2048;
  for (int c = threadIdx.x; c < 2048; c += 256)
    yr[c] = __float2bfloat16(xr[c] * rn * w[c]);
}

// ---------------------------------------------------------------------------
// Generic batched WMMA GEMM:  C[row,col] = scale * sum_k A[row,k] * Bt[col,k]
// A: bf16, lda row stride; Bt: bf16 [N x K] (weights in (out,in) form).
// C stored as bf16 at Cb[row*ldc_row + col*ldc_col] (allows transposed out).
// Block = 256 threads (8 waves) covering a 128(M) x 128(N) tile;
// each wave computes 32x64 (2 A-frags reused across 4 B-frags:
// 12 b128 loads per 8 WMMA). All loads use saddr + 32-bit offsets.
// If beta_ptr != nullptr, scale = sigmoid(*beta).
// ---------------------------------------------------------------------------
__global__ void gemm_bf16_tn_kernel(const __hip_bfloat16* __restrict__ A,
                                    const __hip_bfloat16* __restrict__ Bt,
                                    __hip_bfloat16* __restrict__ C,
                                    int K, int lda, int ldb,
                                    int ldc_row, int ldc_col,
                                    int batchA, int batchB, int batchC,
                                    const float* beta_ptr) {
  const int wave = threadIdx.x >> 5;
  const int lane = threadIdx.x & 31;
  const unsigned lh = lane & 15;
  const unsigned koff = (lane < 16) ? 0u : 8u;
  const unsigned rowadd = (lane < 16) ? 0u : 8u;
  const int mi = wave & 3;   // M sub-tile (32 rows) within 128
  const int nh = wave >> 2;  // N half (64 cols) within 128

  const unsigned b = blockIdx.z;
  const __hip_bfloat16* Ab = A + (unsigned)(b * batchA);
  const __hip_bfloat16* Bb = Bt + (unsigned)(b * batchB);
  __hip_bfloat16* Cb = C + (unsigned)(b * batchC);

  const unsigned mbase = blockIdx.x * 128 + mi * 32;
  const unsigned nbase = blockIdx.y * 128 + nh * 64;

  float scale = 1.0f;
  if (beta_ptr) scale = 1.0f / (1.0f + __expf(-beta_ptr[0]));

  v8f acc[2][4];
#pragma unroll
  for (int si = 0; si < 2; ++si)
#pragma unroll
    for (int t = 0; t < 4; ++t) acc[si][t] = zero8();

  // 32-bit per-lane base offsets (elements); kk advances them uniformly.
  const unsigned aoff0 = (mbase + lh) * (unsigned)lda + koff;
  const unsigned aoff1 = (mbase + 16 + lh) * (unsigned)lda + koff;
  unsigned boff[4];
#pragma unroll
  for (int t = 0; t < 4; ++t)
    boff[t] = (nbase + t * 16 + lh) * (unsigned)ldb + koff;

  for (unsigned kk = 0; kk < (unsigned)K; kk += 32) {
    v16bf af0 = load_frag_u(Ab, aoff0 + kk);
    v16bf af1 = load_frag_u(Ab, aoff1 + kk);
#pragma unroll
    for (int t = 0; t < 4; ++t) {
      v16bf bf = load_frag_u(Bb, boff[t] + kk);
      acc[0][t] = wmma_bf16(af0, bf, acc[0][t]);
      acc[1][t] = wmma_bf16(af1, bf, acc[1][t]);
    }
  }

#pragma unroll
  for (int si = 0; si < 2; ++si) {
#pragma unroll
    for (int t = 0; t < 4; ++t) {
      const unsigned nc = nbase + t * 16 + lh;
#pragma unroll
      for (int r = 0; r < 8; ++r) {
        const unsigned row = mbase + si * 16 + r + rowadd;
        Cb[row * (unsigned)ldc_row + nc * (unsigned)ldc_col] =
            __float2bfloat16(acc[si][t][r] * scale);
      }
    }
  }
}

// ---------------------------------------------------------------------------
// Fused attention + second RMSNorm + residual.
// Block = 512 threads = 16 waves; wave h handles head h for TWO 16-row query
// tiles (32 rows total) so every k/v fragment is reused across both tiles
// (~1 b128 load per WMMA). All global loads use saddr + 32-bit offsets.
// q:  [8192][2048] bf16 (pre-scaled by sigmoid(beta))
// k:  [16*1024][128] bf16
// vT: [16][128][1024] bf16
// Dynamic LDS: 32*2048 f32 rowbuf | 16 waves * 512 bf16 P scratch | 32 f32
// = 278656 B (within the WGP's 320 KB LDS).
// ---------------------------------------------------------------------------
__global__ void attn_fused_kernel(const __hip_bfloat16* __restrict__ qb,
                                  const __hip_bfloat16* __restrict__ kb,
                                  const __hip_bfloat16* __restrict__ vtb,
                                  const float* __restrict__ qin,
                                  const float* __restrict__ nrw,
                                  float* __restrict__ out) {
  extern __shared__ char smem[];
  float* rowbuf = (float*)smem;                                    // 256 KB
  __hip_bfloat16* pbase = (__hip_bfloat16*)(smem + 32 * 2048 * 4); // 16 KB
  float* rnorm = (float*)(smem + 32 * 2048 * 4 + 16 * 512 * 2);    // 128 B

  const int wave = threadIdx.x >> 5;
  const int lane = threadIdx.x & 31;
  const unsigned lh = lane & 15;
  const unsigned koff = (lane < 16) ? 0u : 8u;
  const unsigned rowadd = (lane < 16) ? 0u : 8u;
  const unsigned h = wave;
  const unsigned s0 = blockIdx.x * 32;
  __hip_bfloat16* P = pbase + wave * 512;  // 16 rows x 32 cols bf16

  // Resident q fragments for two 16-row tiles (K = D = 128 -> 4 frags each).
  v16bf qf[2][4];
#pragma unroll
  for (int st = 0; st < 2; ++st) {
    const unsigned qoff = (s0 + st * 16 + lh) * 2048u + h * 128u + koff;
#pragma unroll
    for (int kk = 0; kk < 4; ++kk)
      qf[st][kk] = load_frag_u(qb, qoff + kk * 32u);
  }

  v8f acc[2][8];
  float mrow[2][8], lrow[2][8];
#pragma unroll
  for (int st = 0; st < 2; ++st) {
#pragma unroll
    for (int t = 0; t < 8; ++t) acc[st][t] = zero8();
#pragma unroll
    for (int r = 0; r < 8; ++r) { mrow[st][r] = -1e30f; lrow[st][r] = 0.0f; }
  }

  const unsigned hK = h * (1024u * 128u);  // element offset of head in k
  const unsigned hV = h * (128u * 1024u);  // element offset of head in vT

  for (unsigned mc = 0; mc < 1024; mc += 32) {
    // scores: two 16x32 chunks, contraction over D=128; k frags reused x2
    v8f sc[2][2];
    sc[0][0] = zero8(); sc[0][1] = zero8();
    sc[1][0] = zero8(); sc[1][1] = zero8();
#pragma unroll
    for (int nt = 0; nt < 2; ++nt) {
      const unsigned kbo = hK + (mc + nt * 16 + lh) * 128u + koff;
#pragma unroll
      for (int kk = 0; kk < 4; ++kk) {
        v16bf kf = load_frag_u(kb, kbo + kk * 32u);
        sc[0][nt] = wmma_bf16(qf[0][kk], kf, sc[0][nt]);
        sc[1][nt] = wmma_bf16(qf[1][kk], kf, sc[1][nt]);
      }
    }

    // online softmax per tile; P reshaped C-layout -> A-layout via LDS
    v16bf pf[2];
#pragma unroll
    for (int st = 0; st < 2; ++st) {
      float pscale[8];
#pragma unroll
      for (int r = 0; r < 8; ++r) {
        float mx = fmaxf(sc[st][0][r], sc[st][1][r]);
#pragma unroll
        for (int d = 8; d >= 1; d >>= 1) mx = fmaxf(mx, __shfl_xor(mx, d, 32));
        float mn = fmaxf(mrow[st][r], mx);
        float s = __expf(mrow[st][r] - mn);
        float p0 = __expf(sc[st][0][r] - mn);
        float p1 = __expf(sc[st][1][r] - mn);
        sc[st][0][r] = p0; sc[st][1][r] = p1;
        float ps = p0 + p1;
#pragma unroll
        for (int d = 8; d >= 1; d >>= 1) ps += __shfl_xor(ps, d, 32);
        lrow[st][r] = lrow[st][r] * s + ps;
        mrow[st][r] = mn;
        pscale[r] = s;
      }
#pragma unroll
      for (int t = 0; t < 8; ++t)
#pragma unroll
        for (int r = 0; r < 8; ++r) acc[st][t][r] *= pscale[r];

#pragma unroll
      for (int r = 0; r < 8; ++r) {
        const unsigned row = r + rowadd;
        P[row * 32 + lh]      = __float2bfloat16(sc[st][0][r]);
        P[row * 32 + 16 + lh] = __float2bfloat16(sc[st][1][r]);
      }
      asm volatile("s_wait_dscnt 0" ::: "memory");  // LDS in-order per wave
      const __hip_bfloat16* pp = P + lh * 32 + koff;
      pf[st] = load_frag(pp, pp + 16);
    }

    // acc += P(16x32) * v(32x128); each v frag reused for both tiles
#pragma unroll
    for (int dt = 0; dt < 8; ++dt) {
      const unsigned vbo = hV + (dt * 16 + lh) * 1024u + mc + koff;
      v16bf vf = load_frag_u(vtb, vbo);
      acc[0][dt] = wmma_bf16(pf[0], vf, acc[0][dt]);
      acc[1][dt] = wmma_bf16(pf[1], vf, acc[1][dt]);
    }
  }

  // stage normalized attention rows into the block-wide row buffer
#pragma unroll
  for (int st = 0; st < 2; ++st) {
#pragma unroll
    for (int t = 0; t < 8; ++t) {
#pragma unroll
      for (int r = 0; r < 8; ++r) {
        const unsigned row = st * 16 + r + rowadd;
        rowbuf[row * 2048 + h * 128 + t * 16 + lh] = acc[st][t][r] / lrow[st][r];
      }
    }
  }
  __syncthreads();

  // second RMSNorm: wave w reduces rows w and w+16 (2048 elements each)
#pragma unroll
  for (int rr = 0; rr < 2; ++rr) {
    const int row = wave + rr * 16;
    float ss = 0.0f;
    for (int c = lane; c < 2048; c += 32) {
      float v = rowbuf[row * 2048 + c];
      ss += v * v;
    }
#pragma unroll
    for (int d = 16; d >= 1; d >>= 1) ss += __shfl_xor(ss, d, 32);
    if (lane == 0) rnorm[row] = rsqrtf(ss * (1.0f / 2048.0f) + 1e-6f);
  }
  __syncthreads();

  // apply norm weight + residual, write final output
  for (unsigned i = threadIdx.x; i < 32 * 2048; i += 512) {
    const unsigned row = i >> 11;
    const unsigned col = i & 2047;
    const unsigned g = (s0 + row) * 2048u + col;
    out[g] = rowbuf[i] * rnorm[row] * nrw[col] + qin[g];
  }
}

// ---------------------------------------------------------------------------
// Host-side orchestration
// ---------------------------------------------------------------------------
extern "C" void kernel_launch(void* const* d_in, const int* in_sizes, int n_in,
                              void* d_out, int out_size, void* d_ws,
                              size_t ws_size, hipStream_t stream) {
  (void)in_sizes; (void)n_in; (void)out_size; (void)ws_size;
  const float* q_in   = (const float*)d_in[0];  // [2,4096,2048]
  const float* W_in   = (const float*)d_in[1];  // [2048,2048]
  const float* W_q    = (const float*)d_in[2];  // [128,128]
  const float* W_k    = (const float*)d_in[3];  // [128,128]
  const float* W_v    = (const float*)d_in[4];  // [128,128]
  const float* stored = (const float*)d_in[5];  // [16,1024,128]
  const float* nqw    = (const float*)d_in[6];  // [2048]
  const float* nrw    = (const float*)d_in[7];  // [2048]
  const float* beta   = (const float*)d_in[8];  // scalar
  float* out = (float*)d_out;

  char* ws = (char*)d_ws;
  const size_t oNormed = 0;                                  // reused as q
  const size_t oProj   = oNormed + (size_t)8192 * 2048 * 2;
  const size_t oWin    = oProj   + (size_t)8192 * 2048 * 2;
  const size_t oWq     = oWin    + (size_t)2048 * 2048 * 2;
  const size_t oWk     = oWq     + (size_t)128 * 128 * 2;
  const size_t oWv     = oWk     + (size_t)128 * 128 * 2;
  const size_t oStored = oWv     + (size_t)128 * 128 * 2;
  const size_t oK      = oStored + (size_t)16 * 1024 * 128 * 2;
  const size_t oVT     = oK      + (size_t)16 * 1024 * 128 * 2;

  __hip_bfloat16* nrm_bf = (__hip_bfloat16*)(ws + oNormed);  // then q
  __hip_bfloat16* proj_bf = (__hip_bfloat16*)(ws + oProj);
  __hip_bfloat16* win_bf = (__hip_bfloat16*)(ws + oWin);
  __hip_bfloat16* wq_bf = (__hip_bfloat16*)(ws + oWq);
  __hip_bfloat16* wk_bf = (__hip_bfloat16*)(ws + oWk);
  __hip_bfloat16* wv_bf = (__hip_bfloat16*)(ws + oWv);
  __hip_bfloat16* st_bf = (__hip_bfloat16*)(ws + oStored);
  __hip_bfloat16* k_bf = (__hip_bfloat16*)(ws + oK);
  __hip_bfloat16* vt_bf = (__hip_bfloat16*)(ws + oVT);

  // 1) weight / pattern casts
  cast_bf16_kernel<<<(2048L * 2048 + 255) / 256, 256, 0, stream>>>(W_in, win_bf, 2048L * 2048);
  cast_bf16_kernel<<<(128L * 128 + 255) / 256, 256, 0, stream>>>(W_q, wq_bf, 128L * 128);
  cast_bf16_kernel<<<(128L * 128 + 255) / 256, 256, 0, stream>>>(W_k, wk_bf, 128L * 128);
  cast_bf16_kernel<<<(128L * 128 + 255) / 256, 256, 0, stream>>>(W_v, wv_bf, 128L * 128);
  cast_bf16_kernel<<<(16L * 1024 * 128 + 255) / 256, 256, 0, stream>>>(stored, st_bf, 16L * 1024 * 128);

  // 2) RMSNorm(query) -> bf16
  rmsnorm_cast_kernel<<<8192, 256, 0, stream>>>(q_in, nqw, nrm_bf);

  // 3) proj = normed @ W_in^T   [8192 x 2048 x 2048]
  gemm_bf16_tn_kernel<<<dim3(8192 / 128, 2048 / 128, 1), 256, 0, stream>>>(
      nrm_bf, win_bf, proj_bf, 2048, 2048, 2048, 2048, 1, 0, 0, 0, nullptr);

  // 4) q[h] = proj[:, h*128:(h+1)*128] @ W_q^T, scaled by sigmoid(beta)
  //    (writes over nrm_bf, which is no longer needed)
  gemm_bf16_tn_kernel<<<dim3(8192 / 128, 1, 16), 256, 0, stream>>>(
      proj_bf, wq_bf, nrm_bf, 128, 2048, 128, 2048, 1, 128, 0, 128, beta);

  // 5) k = stored @ W_k^T   [16384 x 128 x 128]
  gemm_bf16_tn_kernel<<<dim3(16384 / 128, 1, 1), 256, 0, stream>>>(
      st_bf, wk_bf, k_bf, 128, 128, 128, 128, 1, 0, 0, 0, nullptr);

  // 6) vT[h][d][m] = (stored[h] @ W_v^T)^T  (batched, transposed store)
  gemm_bf16_tn_kernel<<<dim3(1024 / 128, 1, 16), 256, 0, stream>>>(
      st_bf, wv_bf, vt_bf, 128, 128, 128, 1, 1024,
      1024 * 128, 0, 128 * 1024, nullptr);

  // 7) fused attention + RMSNorm + residual (two 16-row tiles per wave)
  const size_t shbytes = 32 * 2048 * 4 + 16 * 512 * 2 + 128;  // 278656 B LDS
  attn_fused_kernel<<<dim3(8192 / 32), 512, shbytes, stream>>>(
      nrm_bf, k_bf, vt_bf, q_in, nrw, out);
}
